// mambaUNet_30915174596705
// MI455X (gfx1250) — matrix operand order repaired and checked by reading, not compiled, not run on hardware
//
#include <hip/hip_runtime.h>
#include <math.h>

// ---------------------------------------------------------------------------
// Types / constants
// ---------------------------------------------------------------------------
typedef __bf16 bf16_t;
typedef __attribute__((ext_vector_type(16))) __bf16 v16bf;
typedef __attribute__((ext_vector_type(8)))  float  v8f;
typedef __attribute__((ext_vector_type(4)))  unsigned v4u;
typedef __attribute__((ext_vector_type(4)))  int      v4i;
typedef __attribute__((ext_vector_type(8)))  int      v8i;

#define TPB 256
enum { FB = 1, FSP = 2, FACC = 4, FSILU = 8 };

#if defined(__HIP_DEVICE_COMPILE__) && __has_builtin(__builtin_amdgcn_tensor_load_to_lds)
#define GEMM_TDM 1
#else
#define GEMM_TDM 0
#endif

__device__ __forceinline__ float dev_silu(float v) { return v / (1.f + __expf(-v)); }

union U16 { unsigned u[8]; v16bf v; };

// ---------------------------------------------------------------------------
// WMMA GEMM:  C[M,N] = A[M,K](bf16) x B[K,N](bf16)  (+bias, softplus/silu, acc)
// Guarantees from operand builders: K%32==0 (zero-padded), lda%8==0,
// ldb%32==0, N%64==0.  Block tile 64x64, 8 waves (4x2), two 16x16 WMMA
// tiles per wave per K step (A reuse).
//
// A tile staging: Tensor Data Mover (tensor_load_to_lds) when available —
//   2D tile 32(K) x 64(M rows), tensor_dim1 = M-m0 so hardware zero-fills
//   M-partial tiles; D# LDS padding (1 dword per 16) de-conflicts banks.
//   Fallback: clamped-row uint4 loads (no exec-mask branches).
// B tile staged transposed (N x K) by all threads so fragment gathers are
// contiguous dword loads.
// ---------------------------------------------------------------------------
__global__ __launch_bounds__(256)
void k_gemm_bf16(const bf16_t* __restrict__ A, const bf16_t* __restrict__ Bm,
                 float* __restrict__ C,
                 int M, int N, int K, int lda, int ldb, int ldc,
                 long sA, long sB, long sC, int aMod,
                 const float* __restrict__ bias, long sBias, int biasMod,
                 int flags)
{
#if GEMM_TDM
    __shared__ bf16_t As[64][34];   // 68B rows: TDM pads 1 dword per 16 dwords
#else
    __shared__ bf16_t As[64][40];   // 80B rows -> 16B-aligned vector stores
#endif
    __shared__ bf16_t Bt[64][40];   // N x K tile (transposed)

    const int bz   = blockIdx.z;
    const int aIdx = aMod ? (bz % aMod) : bz;
    const bf16_t* Ab = A + (long)aIdx * sA;
    const bf16_t* Bb = Bm + (long)bz * sB;
    float*        Cb = C + (long)bz * sC;
    const float*  biasb = bias ? (bias + (long)(biasMod ? (bz % biasMod) : bz) * sBias)
                               : (const float*)0;

    const int tid  = threadIdx.x;
    const int lane = tid & 31;
    const int wave = tid >> 5;
    const int wm   = wave >> 1;        // 0..3
    const int wn   = wave & 1;         // 0..1
    const int m0   = blockIdx.y * 64;
    const int n0   = blockIdx.x * 64;

    const int br = tid >> 3, bc = (tid & 7) * 8;   // B: one uint4 per thread
#if !GEMM_TDM
    const int ar = tid >> 2, ac = (tid & 3) * 8;   // A: one uint4 per thread
    int aRow = m0 + ar; if (aRow > M - 1) aRow = M - 1;   // branch-free clamp
#endif

    v8f acc0 = {};
    v8f acc1 = {};

    for (int k0 = 0; k0 < K; k0 += 32) {
#if GEMM_TDM
        // ---- stage A (64 rows x 32 K) via Tensor Data Mover ----
        if (wave == 0) {
            unsigned long long ga =
                (unsigned long long)(const void*)(Ab + (long)m0 * lda + k0);
            unsigned ldsA = (unsigned)(unsigned long long)(const void*)&As[0][0];
            unsigned remM = (unsigned)(M - m0); if (remM > 64u) remM = 64u;
            // D# group0: count=1 | lds_addr | global_addr[56:0] | type=2
            v4u g0;
            g0.x = 1u;
            g0.y = ldsA;
            g0.z = (unsigned)(ga & 0xFFFFFFFFull);
            g0.w = (unsigned)((ga >> 32) & 0x01FFFFFFull) | (2u << 30);
            // D# group1: data_size=2B | pad 1 dword per 16 | dims/strides
            unsigned long long st0 = (unsigned long long)lda;  // elements
            v8i g1;
            g1[0] = (int)((1u << 16) | (1u << 20) | (3u << 22));
            g1[1] = (int)((32u & 0xFFFFu) << 16);              // tensor_dim0 = 32
            g1[2] = (int)(remM << 16);                         // tensor_dim1 = M-m0
            g1[3] = (int)((32u) << 16);                        // tile_dim0 = 32
            g1[4] = (int)(64u);                                // tile_dim1 = 64
            g1[5] = (int)(st0 & 0xFFFFFFFFull);                // dim0 stride lo
            g1[6] = (int)((st0 >> 32) & 0xFFFFull);            // dim0 stride hi
            g1[7] = 0;
            v4i g2 = {0, 0, 0, 0};
            v4i g3 = {0, 0, 0, 0};
#if __clang_major__ >= 23
            v8i g4 = {0, 0, 0, 0, 0, 0, 0, 0};
            __builtin_amdgcn_tensor_load_to_lds(g0, g1, g2, g3, g4, 0);
#else
            __builtin_amdgcn_tensor_load_to_lds(g0, g1, g2, g3, 0);
#endif
            __builtin_amdgcn_s_wait_tensorcnt(0);
        }
#else
        // ---- stage A (64x32): clamped vector load, no branches ----
        *(uint4*)&As[ar][ac] =
            *(const uint4*)(Ab + (long)aRow * lda + (k0 + ac));
        if (k0 + 32 < K)
            __builtin_prefetch(Ab + (long)aRow * lda + (k0 + 32 + ac), 0, 1);
#endif
        // ---- stage B (32x64) transposed into Bt (64x40) ----
        {
            bf16_t tmp[8];
            *(uint4*)tmp = *(const uint4*)(Bb + (long)(k0 + br) * ldb + (n0 + bc));
            #pragma unroll
            for (int i = 0; i < 8; ++i) Bt[bc + i][br] = tmp[i];
        }
        if (k0 + 32 < K)
            __builtin_prefetch(Bb + (long)(k0 + 32 + br) * ldb + (n0 + bc), 0, 1);
        __syncthreads();

        // ---- fragments (CDNA5 16-bit layouts, CDNA5 ISA): dword gathers ----
        const int rc   = lane & 15;
        const int half = lane >> 4;
        const bf16_t* arow  = &As[wm * 16 + rc][0];
        const bf16_t* b0row = &Bt[wn * 32 + rc][0];
        const bf16_t* b1row = &Bt[wn * 32 + 16 + rc][0];
        U16 af, bf0, bf1;
        #pragma unroll
        for (int v2 = 0; v2 < 8; ++v2) {
            int baseK = ((v2 < 4) ? 0 : 16) + half * 8 + (v2 & 3) * 2;
            int kk    = baseK;
            af.u[v2]  = *(const unsigned*)(arow + baseK);
            bf0.u[v2] = *(const unsigned*)(b0row + kk);
            bf1.u[v2] = *(const unsigned*)(b1row + kk);
        }
        acc0 = __builtin_amdgcn_wmma_f32_16x16x32_bf16(
                   false, af.v, false, bf0.v, (short)0, acc0, false, false);
        acc1 = __builtin_amdgcn_wmma_f32_16x16x32_bf16(
                   false, af.v, false, bf1.v, (short)0, acc1, false, false);
        __syncthreads();
    }

    // ---- epilogue: C layout lane = N col, VGPR i + 8*(lane>>4) = M row ----
    const int col = lane & 15;
    const int rh  = lane >> 4;
    #pragma unroll
    for (int i = 0; i < 8; ++i) {
        int m = m0 + wm * 16 + rh * 8 + i;
        if (m >= M) continue;
        float v0 = acc0[i], v1 = acc1[i];
        if (flags & FB)    { float bv = biasb[m]; v0 += bv; v1 += bv; }
        if (flags & FSP)   { v0 = (v0 > 20.f) ? v0 : log1pf(__expf(v0));
                             v1 = (v1 > 20.f) ? v1 : log1pf(__expf(v1)); }
        if (flags & FSILU) { v0 = dev_silu(v0); v1 = dev_silu(v1); }
        long row = (long)m * ldc;
        int na  = n0 + wn * 32 + col;
        int nb2 = na + 16;
        if (na < N)  { if (flags & FACC) Cb[row + na]  += v0; else Cb[row + na]  = v0; }
        if (nb2 < N) { if (flags & FACC) Cb[row + nb2] += v1; else Cb[row + nb2] = v1; }
    }
}

// ---------------------------------------------------------------------------
// Support kernels
// ---------------------------------------------------------------------------
__global__ void k_cast_bf16(const float* __restrict__ s, bf16_t* __restrict__ d, long n) {
    long i = (long)blockIdx.x * blockDim.x + threadIdx.x;
    if (i < n) d[i] = (bf16_t)s[i];
}

// cast (nb, rows, cs) f32 -> (nb, rows, cd) bf16, zero-padding cols >= cs
__global__ void k_cast2d(const float* __restrict__ s, bf16_t* __restrict__ d,
                         int rows, int cs, int cd, long sB, long dB, int nb) {
    long per = (long)rows * cd;
    long total = per * nb;
    long i = (long)blockIdx.x * blockDim.x + threadIdx.x;
    if (i >= total) return;
    int  b  = (int)(i / per);
    long rr = i % per;
    int row = (int)(rr / cd);
    int col = (int)(rr % cd);
    float v = (col < cs) ? s[(long)b * sB + (long)row * cs + col] : 0.f;
    d[(long)b * dB + rr] = (bf16_t)v;
}

// cast (nb, R, Lc) f32 -> (nb, RP, Lc) bf16, zero-padding rows >= R
__global__ void k_cast_pad_rows(const float* __restrict__ s, bf16_t* __restrict__ d,
                                int R, int RP, long Lc, long sB, long dB, int nb) {
    long per = (long)RP * Lc;
    long total = per * nb;
    long i = (long)blockIdx.x * blockDim.x + threadIdx.x;
    if (i >= total) return;
    int  b  = (int)(i / per);
    long rr = i % per;
    int  row = (int)(rr / Lc);
    long col = rr % Lc;
    float v = (row < R) ? s[(long)b * sB + (long)row * Lc + col] : 0.f;
    d[(long)b * dB + rr] = (bf16_t)v;
}

__global__ void k_copy_f32(const float* __restrict__ s, float* __restrict__ d, long n) {
    long i = (long)blockIdx.x * blockDim.x + threadIdx.x;
    if (i < n) d[i] = s[i];
}

__global__ void k_negexp(const float* __restrict__ s, float* __restrict__ d, long n) {
    long i = (long)blockIdx.x * blockDim.x + threadIdx.x;
    if (i < n) d[i] = -__expf(s[i]);
}

// im2col fused with f32->bf16 cast, K-dim padded: col (B, KP, OH*OW)
__global__ void k_im2col_bf16(const float* __restrict__ x, bf16_t* __restrict__ col,
                              int C, int H, int W, int KH, int KW,
                              int stride, int pad, int OH, int OW, int B,
                              int Kc, int KP) {
    long L = (long)OH * OW;
    long total = (long)B * KP * L;
    long i = (long)blockIdx.x * blockDim.x + threadIdx.x;
    if (i >= total) return;
    long l = i % L;  long t = i / L;
    int kidx = (int)(t % KP);  int b = (int)(t / KP);
    float v = 0.f;
    if (kidx < Kc) {
        int ow = (int)(l % OW), oh = (int)(l / OW);
        int kw = kidx % KW;  int t2 = kidx / KW;
        int kh = t2 % KH;    int c  = t2 / KH;
        int ih = oh * stride - pad + kh;
        int iw = ow * stride - pad + kw;
        if (ih >= 0 && ih < H && iw >= 0 && iw < W)
            v = x[(((long)b * C + c) * H + ih) * W + iw];
    }
    col[i] = (bf16_t)v;
}

// GroupNorm(8 groups) + SiLU. grid = (8, B)
__global__ __launch_bounds__(256)
void k_gn_silu(const float* __restrict__ x, float* __restrict__ y,
               const float* __restrict__ g, const float* __restrict__ bta,
               int C, long HW) {
    const int grp = blockIdx.x;
    const int bb  = blockIdx.y;
    const int cg  = C / 8;
    const long n  = (long)cg * HW;
    const float* xb = x + ((long)bb * C + (long)grp * cg) * HW;
    float*       yb = y + ((long)bb * C + (long)grp * cg) * HW;

    float s = 0.f, s2 = 0.f;
    for (long i = threadIdx.x; i < n; i += blockDim.x) {
        float v = xb[i]; s += v; s2 += v * v;
    }
    __shared__ float rs[256], rq[256];
    rs[threadIdx.x] = s; rq[threadIdx.x] = s2;
    __syncthreads();
    for (int o = 128; o > 0; o >>= 1) {
        if (threadIdx.x < o) { rs[threadIdx.x] += rs[threadIdx.x + o]; rq[threadIdx.x] += rq[threadIdx.x + o]; }
        __syncthreads();
    }
    float mu  = rs[0] / (float)n;
    float var = rq[0] / (float)n - mu * mu;
    float inv = rsqrtf(var + 1e-5f);
    for (long i = threadIdx.x; i < n; i += blockDim.x) {
        int c = grp * cg + (int)(i / HW);
        float v = (xb[i] - mu) * inv * g[c] + bta[c];
        yb[i] = dev_silu(v);
    }
}

// LayerNorm over R strided rows (channel dim); thread per column (coalesced).
__global__ void k_ln_cols(const float* __restrict__ x, float* __restrict__ y,
                          const float* __restrict__ g, const float* __restrict__ bta,
                          int R, long cols, long rowStride, long bStride, int nb) {
    long total = cols * nb;
    long i = (long)blockIdx.x * blockDim.x + threadIdx.x;
    if (i >= total) return;
    int  bb = (int)(i / cols);
    long l  = i % cols;
    const float* xb = x + (long)bb * bStride + l;
    float*       yb = y + (long)bb * bStride + l;
    float s = 0.f, s2 = 0.f;
    for (int r = 0; r < R; ++r) { float v = xb[(long)r * rowStride]; s += v; s2 += v * v; }
    float mu  = s / (float)R;
    float var = s2 / (float)R - mu * mu;
    float inv = rsqrtf(var + 1e-5f);
    for (int r = 0; r < R; ++r)
        yb[(long)r * rowStride] = (xb[(long)r * rowStride] - mu) * inv * g[r] + bta[r];
}

// depthwise 3x3 conv + SiLU; input rows live inside a larger tensor (inBStride)
__global__ void k_dwconv3_silu(const float* __restrict__ x, long inBStride,
                               const float* __restrict__ w, const float* __restrict__ bias,
                               float* __restrict__ y, int d, int H, int W, int B) {
    long HW = (long)H * W;
    long total = (long)B * d * HW;
    long i = (long)blockIdx.x * blockDim.x + threadIdx.x;
    if (i >= total) return;
    int wc = (int)(i % W);  long t = i / W;
    int h  = (int)(t % H);  t /= H;
    int c  = (int)(t % d);  int b = (int)(t / d);
    const float* xb = x + (long)b * inBStride + (long)c * HW;
    const float* wp = w + (long)c * 9;
    float acc = bias[c];
    #pragma unroll
    for (int kh = 0; kh < 3; ++kh) {
        int ih = h - 1 + kh;
        if (ih < 0 || ih >= H) continue;
        #pragma unroll
        for (int kw = 0; kw < 3; ++kw) {
            int iw = wc - 1 + kw;
            if (iw < 0 || iw >= W) continue;
            acc += xb[(long)ih * W + iw] * wp[kh * 3 + kw];
        }
    }
    y[i] = dev_silu(acc);
}

// build the 4 scan directions xs (B,4,d,L) from xx (B,d,H,W)
__global__ void k_build_xs(const float* __restrict__ xx, float* __restrict__ xs,
                           int d, int H, int W, int B) {
    long L = (long)H * W;
    long total = (long)B * d * L;
    long i = (long)blockIdx.x * blockDim.x + threadIdx.x;
    if (i >= total) return;
    int wc = (int)(i % W);  long t = i / W;
    int h  = (int)(t % H);  t /= H;
    int c  = (int)(t % d);  int b = (int)(t / d);
    float v = xx[i];
    long l  = (long)h * W + wc;
    long lt = (long)wc * H + h;
    long b4 = (long)b * 4;
    xs[((b4 + 0) * d + c) * L + l]            = v;
    xs[((b4 + 1) * d + c) * L + lt]           = v;
    xs[((b4 + 2) * d + c) * L + (L - 1 - l)]  = v;
    xs[((b4 + 3) * d + c) * L + (L - 1 - lt)] = v;
}

// selective scan: one thread per (b,k,d) row, N=16 state in registers
__global__ void k_scan(const float* __restrict__ u, const float* __restrict__ dt,
                       const float* __restrict__ Aexp, const float* __restrict__ xdbl,
                       const float* __restrict__ Dp, float* __restrict__ ys,
                       int d, long L, int Cp, int r, int B) {
    long total = (long)B * 4 * d;
    long i = (long)blockIdx.x * blockDim.x + threadIdx.x;
    if (i >= total) return;
    int c = (int)(i % d);  long t = i / d;
    int k = (int)(t % 4);  int b = (int)(t / 4);
    long rowOff = (((long)b * 4 + k) * d + c) * L;
    const float* ub  = u  + rowOff;
    const float* dtb = dt + rowOff;
    const float* Bsp = xdbl + (((long)b * 4 + k) * Cp + r) * L;
    const float* Csp = Bsp + 16 * L;
    const float* Ar  = Aexp + ((long)k * d + c) * 16;
    float Dv = Dp[(long)k * d + c];
    float h[16], a[16];
    #pragma unroll
    for (int n = 0; n < 16; ++n) { h[n] = 0.f; a[n] = Ar[n]; }
    float* yb = ys + rowOff;
    for (long l = 0; l < L; ++l) {
        float dtv = dtb[l], uv = ub[l];
        float du = dtv * uv;
        float y = 0.f;
        #pragma unroll
        for (int n = 0; n < 16; ++n) {
            float dA = __expf(dtv * a[n]);
            h[n] = dA * h[n] + du * Bsp[(long)n * L + l];
            y += h[n] * Csp[(long)n * L + l];
        }
        yb[l] = y + Dv * uv;
    }
}

// combine 4 scan directions back: y (B,d,L)
__global__ void k_combine_y(const float* __restrict__ ys, float* __restrict__ y,
                            int d, int H, int W, int B) {
    long L = (long)H * W;
    long total = (long)B * d * L;
    long i = (long)blockIdx.x * blockDim.x + threadIdx.x;
    if (i >= total) return;
    int wc = (int)(i % W);  long t = i / W;
    int h  = (int)(t % H);  t /= H;
    int c  = (int)(t % d);  int b = (int)(t / d);
    long l  = (long)h * W + wc;
    long lt = (long)wc * H + h;
    long b4 = (long)b * 4;
    const float* y0 = ys + ((b4 + 0) * d + c) * L;
    const float* y1 = ys + ((b4 + 1) * d + c) * L;
    const float* y2 = ys + ((b4 + 2) * d + c) * L;
    const float* y3 = ys + ((b4 + 3) * d + c) * L;
    y[((long)b * d + c) * L + l] = y0[l] + y2[L - 1 - l] + y1[lt] + y3[L - 1 - lt];
}

// y *= silu(z) where z = rows [d,2d) of xz (B,2d,L)
__global__ void k_gate(float* __restrict__ y, const float* __restrict__ xz,
                       int d, long L, int B) {
    long per = (long)d * L;
    long total = per * B;
    long i = (long)blockIdx.x * blockDim.x + threadIdx.x;
    if (i >= total) return;
    int  b   = (int)(i / per);
    long rem = i % per;
    float zv = xz[((long)b * 2 * d + d) * L + rem];
    y[i] *= dev_silu(zv);
}

__global__ void k_upsample2(const float* __restrict__ x, float* __restrict__ y,
                            int C, int H, int W, int B) {
    int H2 = 2 * H, W2 = 2 * W;
    long total = (long)B * C * H2 * W2;
    long i = (long)blockIdx.x * blockDim.x + threadIdx.x;
    if (i >= total) return;
    int ow = (int)(i % W2);  long t = i / W2;
    int oh = (int)(t % H2);  t /= H2;
    int c  = (int)(t % C);   int b = (int)(t / C);
    y[i] = x[(((long)b * C + c) * H + (oh >> 1)) * W + (ow >> 1)];
}

// channel-slice copy (for concat)
__global__ void k_copy_ch(const float* __restrict__ src, float* __restrict__ dst,
                          int B, int Csrc, int Cdst, int c0, long L) {
    long total = (long)B * Csrc * L;
    long i = (long)blockIdx.x * blockDim.x + threadIdx.x;
    if (i >= total) return;
    long l = i % L;  long t = i / L;
    int c = (int)(t % Csrc);  int b = (int)(t / Csrc);
    dst[((long)b * Cdst + c0 + c) * L + l] = src[i];
}

// ConvTranspose k=2 s=2, weight layout (Cin, Cout, 2, 2)
__global__ void k_convt2(const float* __restrict__ x, const float* __restrict__ w,
                         const float* __restrict__ bias, float* __restrict__ y,
                         int Cin, int Cout, int H, int W, int B, int dosilu) {
    int H2 = 2 * H, W2 = 2 * W;
    long total = (long)B * Cout * H2 * W2;
    long i = (long)blockIdx.x * blockDim.x + threadIdx.x;
    if (i >= total) return;
    int ow = (int)(i % W2);  long t = i / W2;
    int oh = (int)(t % H2);  t /= H2;
    int o  = (int)(t % Cout); int b = (int)(t / Cout);
    int h = oh >> 1, kk = oh & 1, wc = ow >> 1, ll = ow & 1;
    float acc = bias[o];
    for (int c = 0; c < Cin; ++c)
        acc += x[(((long)b * Cin + c) * H + h) * W + wc] *
               w[(((long)c * Cout + o) * 2 + kk) * 2 + ll];
    y[i] = dosilu ? dev_silu(acc) : acc;
}

// ---------------------------------------------------------------------------
// Host side: workspace arena, param decoding, orchestration
// ---------------------------------------------------------------------------
struct WS {
    char*  base;
    size_t lo, hi, cap;
    void* P(size_t b) { size_t o = (lo + 255) & ~(size_t)255; lo = o + b; return base + o; }
    void* T(size_t b) { size_t b2 = (b + 255) & ~(size_t)255; hi -= b2; return base + hi; }
};

static inline unsigned gdu(long n) { return (unsigned)((n + TPB - 1) / TPB); }
static inline long rup32(long k) { return (k + 31) & ~31L; }

struct ResP {
    const float *c1b, *c1w, *c2b, *c2w, *n1b, *n1g, *n2b, *n2g;
    const float *scb, *scw, *snb, *sng;
    bool sc;
};
struct VSSP {
    const float *A_log, *D, *dt_b, *dt_w, *dwc_b, *dwc_w, *in_w;
    const float *lnb, *lng, *onb, *ong, *out_w, *xproj_w;
};
struct LayerP { ResP res; VSSP v1, v2; };
struct StageP {
    ResP in_res; LayerP layers[2]; int depth;
    const float *pool_b, *pool_w, *up_b, *up_w;
};

struct Cur {
    void* const* din; int i;
    const float* next() { return (const float*)din[i++]; }
};

static ResP loadRes(Cur& c, bool sc) {
    ResP p{};
    p.c1b = c.next(); p.c1w = c.next(); p.c2b = c.next(); p.c2w = c.next();
    p.n1b = c.next(); p.n1g = c.next(); p.n2b = c.next(); p.n2g = c.next();
    if (sc) { p.scb = c.next(); p.scw = c.next(); p.snb = c.next(); p.sng = c.next(); }
    p.sc = sc;
    return p;
}
static VSSP loadVSS(Cur& c) {
    VSSP p{};
    p.A_log = c.next(); p.D = c.next(); p.dt_b = c.next(); p.dt_w = c.next();
    p.dwc_b = c.next(); p.dwc_w = c.next(); p.in_w = c.next();
    p.lnb = c.next(); p.lng = c.next(); p.onb = c.next(); p.ong = c.next();
    p.out_w = c.next(); p.xproj_w = c.next();
    return p;
}
static StageP loadStage(Cur& c, bool sc, int depth, bool pool, bool up) {
    StageP s{};
    s.in_res = loadRes(c, sc);
    s.depth = depth;
    for (int i = 0; i < depth; ++i) {
        s.layers[i].res = loadRes(c, false);
        s.layers[i].v1  = loadVSS(c);
        s.layers[i].v2  = loadVSS(c);
    }
    if (pool) { s.pool_b = c.next(); s.pool_w = c.next(); }
    if (up)   { s.up_b = c.next();  s.up_w = c.next(); }
    return s;
}

static void gemm(hipStream_t st, const bf16_t* A, const bf16_t* B, float* C,
                 int M, int N, int K, int lda, int ldb, int ldc,
                 long sA, long sB, long sC, int aMod,
                 const float* bias, long sBias, int biasMod, int flags, int nb) {
    dim3 g((unsigned)((N + 63) / 64), (unsigned)((M + 63) / 64), (unsigned)nb);
    k_gemm_bf16<<<g, TPB, 0, st>>>(A, B, C, M, N, K, lda, ldb, ldc,
                                   sA, sB, sC, aMod, bias, sBias, biasMod, flags);
}

// generic conv (im2col + WMMA GEMM). out preallocated (B,O,OH*OW).
static void convop(WS& ws, hipStream_t st, const float* x, const float* w, const float* bias,
                   float* out, int B, int C, int H, int W, int O, int KS,
                   int stride, int pad, int flags) {
    int OH = (H + 2 * pad - KS) / stride + 1;
    int OW = (W + 2 * pad - KS) / stride + 1;
    long L  = (long)OH * OW;
    long Kc = (long)C * KS * KS;
    long KP = rup32(Kc);
    size_t mark = ws.hi;
    bf16_t* col = (bf16_t*)ws.T((size_t)B * KP * L * sizeof(bf16_t));
    bf16_t* wb  = (bf16_t*)ws.T((size_t)O * KP * sizeof(bf16_t));
    k_im2col_bf16<<<gdu((long)B * KP * L), TPB, 0, st>>>(
        x, col, C, H, W, KS, KS, stride, pad, OH, OW, B, (int)Kc, (int)KP);
    k_cast2d<<<gdu((long)O * KP), TPB, 0, st>>>(w, wb, O, (int)Kc, (int)KP, 0, 0, 1);
    gemm(st, wb, col, out, O, (int)L, (int)KP, (int)KP, (int)L, (int)L,
         0, KP * L, (long)O * L, 1, bias, 0, 1, flags, B);
    ws.hi = mark;
}

static float* resblockF(WS& ws, hipStream_t st, const ResP& p, const float* x,
                        int B, int Cin, int Cout, int H, int W) {
    long L = (long)H * W;
    size_t mark = ws.hi;
    float* t1 = (float*)ws.T((size_t)B * Cin * L * 4);
    k_gn_silu<<<dim3(8, B), TPB, 0, st>>>(x, t1, p.n1g, p.n1b, Cin, L);
    float* h = (float*)ws.T((size_t)B * Cout * L * 4);
    convop(ws, st, t1, p.c1w, p.c1b, h, B, Cin, H, W, Cout, 3, 1, 1, FB);
    float* t2 = (float*)ws.T((size_t)B * Cout * L * 4);
    k_gn_silu<<<dim3(8, B), TPB, 0, st>>>(h, t2, p.n2g, p.n2b, Cout, L);
    float* out = (float*)ws.P((size_t)B * Cout * L * 4);
    if (p.sc) {
        float* ts = (float*)ws.T((size_t)B * Cin * L * 4);
        k_gn_silu<<<dim3(8, B), TPB, 0, st>>>(x, ts, p.sng, p.snb, Cin, L);
        convop(ws, st, ts, p.scw, p.scb, out, B, Cin, H, W, Cout, 1, 1, 0, FB);
    } else {
        k_copy_f32<<<gdu((long)B * Cin * L), TPB, 0, st>>>(x, out, (long)B * Cin * L);
    }
    convop(ws, st, t2, p.c2w, p.c2b, out, B, Cout, H, W, Cout, 3, 1, 1, FB | FACC);
    ws.hi = mark;
    return out;
}

// VSS block: in-place residual update of x (B,C,H*W)
static void vssF(WS& ws, hipStream_t st, const VSSP& p, float* x,
                 int B, int C, int H, int W) {
    const int d  = 2 * C;
    const int r  = (C + 15) / 16;
    const int Cp = r + 32;
    const long L = (long)H * W;
    size_t mark = ws.hi;

    // layer_norm over C
    float* t = (float*)ws.T((size_t)B * C * L * 4);
    k_ln_cols<<<gdu(L * B), TPB, 0, st>>>(x, t, p.lng, p.lnb, C, L, L, (long)C * L, B);
    bf16_t* tb = (bf16_t*)ws.T((size_t)B * C * L * 2);
    k_cast_bf16<<<gdu((long)B * C * L), TPB, 0, st>>>(t, tb, (long)B * C * L);

    // in-projection: xz (B, 2d, L), K = C (multiple of 32)
    bf16_t* inwb = (bf16_t*)ws.T((size_t)2 * d * C * 2);
    k_cast_bf16<<<gdu((long)2 * d * C), TPB, 0, st>>>(p.in_w, inwb, (long)2 * d * C);
    float* xz = (float*)ws.T((size_t)B * 2 * d * L * 4);
    gemm(st, inwb, tb, xz, 2 * d, (int)L, C, C, (int)L, (int)L,
         0, (long)C * L, (long)2 * d * L, 1, 0, 0, 1, 0, B);

    // depthwise conv + silu on xx = rows [0,d)
    float* xx = (float*)ws.T((size_t)B * d * L * 4);
    k_dwconv3_silu<<<gdu((long)B * d * L), TPB, 0, st>>>(xz, (long)2 * d * L,
                                                         p.dwc_w, p.dwc_b, xx, d, H, W, B);

    // 4 scan directions
    float* xs = (float*)ws.T((size_t)B * 4 * d * L * 4);
    k_build_xs<<<gdu((long)B * d * L), TPB, 0, st>>>(xx, xs, d, H, W, B);
    bf16_t* xsb = (bf16_t*)ws.T((size_t)B * 4 * d * L * 2);
    k_cast_bf16<<<gdu((long)B * 4 * d * L), TPB, 0, st>>>(xs, xsb, (long)B * 4 * d * L);

    // x_dbl (B,4,Cp,L) = xproj_w[k] (Cp,d) x xs[b,k] (d,L), K = d (mult of 32)
    bf16_t* xpb = (bf16_t*)ws.T((size_t)4 * Cp * d * 2);
    k_cast_bf16<<<gdu((long)4 * Cp * d), TPB, 0, st>>>(p.xproj_w, xpb, (long)4 * Cp * d);
    float* xdbl = (float*)ws.T((size_t)B * 4 * Cp * L * 4);
    gemm(st, xpb, xsb, xdbl, Cp, (int)L, d, d, (int)L, (int)L,
         (long)Cp * d, (long)d * L, (long)Cp * L, 4, 0, 0, 1, 0, B * 4);

    // dt (B,4,d,L) = dt_w[k] (d,r) x dts (r,L) + dt_b[k], softplus.  K padded to 32.
    bf16_t* dtrb = (bf16_t*)ws.T((size_t)B * 4 * 32 * L * 2);
    k_cast_pad_rows<<<gdu((long)B * 4 * 32 * L), TPB, 0, st>>>(
        xdbl, dtrb, r, 32, L, (long)Cp * L, (long)32 * L, B * 4);
    bf16_t* dtwb = (bf16_t*)ws.T((size_t)4 * d * 32 * 2);
    k_cast2d<<<gdu((long)4 * d * 32), TPB, 0, st>>>(
        p.dt_w, dtwb, d, r, 32, (long)d * r, (long)d * 32, 4);
    float* dt = (float*)ws.T((size_t)B * 4 * d * L * 4);
    gemm(st, dtwb, dtrb, dt, d, (int)L, 32, 32, (int)L, (int)L,
         (long)d * 32, (long)32 * L, (long)d * L, 4, p.dt_b, d, 4, FB | FSP, B * 4);

    // A = -exp(A_log)
    float* Aexp = (float*)ws.T((size_t)4 * d * 16 * 4);
    k_negexp<<<gdu((long)4 * d * 16), TPB, 0, st>>>(p.A_log, Aexp, (long)4 * d * 16);

    // selective scan
    float* ys = (float*)ws.T((size_t)B * 4 * d * L * 4);
    k_scan<<<gdu((long)B * 4 * d), TPB, 0, st>>>(xs, dt, Aexp, xdbl, p.D, ys,
                                                 d, L, Cp, r, B);

    // combine directions, out-norm, gate
    float* y = (float*)ws.T((size_t)B * d * L * 4);
    k_combine_y<<<gdu((long)B * d * L), TPB, 0, st>>>(ys, y, d, H, W, B);
    float* yn = (float*)ws.T((size_t)B * d * L * 4);
    k_ln_cols<<<gdu(L * B), TPB, 0, st>>>(y, yn, p.ong, p.onb, d, L, L, (long)d * L, B);
    k_gate<<<gdu((long)B * d * L), TPB, 0, st>>>(yn, xz, d, L, B);

    // out-projection with residual accumulate into x, K = d
    bf16_t* ynb = (bf16_t*)ws.T((size_t)B * d * L * 2);
    k_cast_bf16<<<gdu((long)B * d * L), TPB, 0, st>>>(yn, ynb, (long)B * d * L);
    bf16_t* owb = (bf16_t*)ws.T((size_t)C * d * 2);
    k_cast_bf16<<<gdu((long)C * d), TPB, 0, st>>>(p.out_w, owb, (long)C * d);
    gemm(st, owb, ynb, x, C, (int)L, d, d, (int)L, (int)L,
         0, (long)d * L, (long)C * L, 1, 0, 0, 1, FACC, B);

    ws.hi = mark;
}

static float* stageBodyF(WS& ws, hipStream_t st, StageP& sp, const float* x,
                         int B, int Cin, int Cout, int H, int W) {
    float* h = resblockF(ws, st, sp.in_res, x, B, Cin, Cout, H, W);
    for (int i = 0; i < sp.depth; ++i) {
        vssF(ws, st, sp.layers[i].v1, h, B, Cout, H, W);
        vssF(ws, st, sp.layers[i].v2, h, B, Cout, H, W);
        h = resblockF(ws, st, sp.layers[i].res, h, B, Cout, Cout, H, W);
    }
    return h;
}

static float* upStageF(WS& ws, hipStream_t st, StageP& sp, const float* m,
                       const float* skip, int B, int Cm, int Cout, int H, int W) {
    int H2 = 2 * H, W2 = 2 * W;
    long L2 = (long)H2 * W2;
    size_t mark = ws.hi;
    float* u = (float*)ws.T((size_t)B * Cm * L2 * 4);
    k_upsample2<<<gdu((long)B * Cm * L2), TPB, 0, st>>>(m, u, Cm, H, W, B);
    float* cv = (float*)ws.T((size_t)B * Cout * L2 * 4);
    convop(ws, st, u, sp.up_w, sp.up_b, cv, B, Cm, H2, W2, Cout, 3, 1, 1, FB);
    float* cat = (float*)ws.T((size_t)B * 2 * Cout * L2 * 4);
    k_copy_ch<<<gdu((long)B * Cout * L2), TPB, 0, st>>>(cv,   cat, B, Cout, 2 * Cout, 0,    L2);
    k_copy_ch<<<gdu((long)B * Cout * L2), TPB, 0, st>>>(skip, cat, B, Cout, 2 * Cout, Cout, L2);
    float* h = stageBodyF(ws, st, sp, cat, B, 2 * Cout, Cout, H2, W2);
    ws.hi = mark;
    return h;
}

// ---------------------------------------------------------------------------
// Entry point
// ---------------------------------------------------------------------------
extern "C" void kernel_launch(void* const* d_in, const int* in_sizes, int n_in,
                              void* d_out, int out_size, void* d_ws, size_t ws_size,
                              hipStream_t stream) {
    (void)in_sizes; (void)n_in; (void)out_size;
    const int B = 2;
    WS ws{(char*)d_ws, 0, ws_size & ~(size_t)255, ws_size};

    // ---- decode params (jax tree-flatten order: sorted dict keys, DFS) ----
    const float* img = (const float*)d_in[0];
    Cur cur{d_in, 1};
    StageP down1 = loadStage(cur, false, 1, true,  false);   // 32 -> 32
    StageP down2 = loadStage(cur, true,  1, true,  false);   // 32 -> 64
    StageP down3 = loadStage(cur, true,  1, true,  false);   // 64 -> 128
    const float* in0_b = cur.next(); const float* in0_w = cur.next();
    const float* in1_b = cur.next(); const float* in1_w = cur.next();
    StageP midS  = loadStage(cur, true,  2, false, false);   // 128 -> 256
    const float* out0_b = cur.next(); const float* out0_w = cur.next();
    const float* out1_b = cur.next(); const float* out1_w = cur.next();
    StageP up1S  = loadStage(cur, true,  1, false, true);    // 64 -> 32
    StageP up2S  = loadStage(cur, true,  1, false, true);    // 128 -> 64
    StageP up3S  = loadStage(cur, true,  1, false, true);    // 256 -> 128

    // ---- stem ----
    float* x  = (float*)ws.P((size_t)B * 16 * 128 * 128 * 4);
    convop(ws, stream, img, in0_w, in0_b, x, B, 1, 256, 256, 16, 3, 2, 1, FB | FSILU);
    float* x0 = (float*)ws.P((size_t)B * 32 * 64 * 64 * 4);
    convop(ws, stream, x, in1_w, in1_b, x0, B, 16, 128, 128, 32, 3, 2, 1, FB);

    // ---- encoder ----
    float* s1 = stageBodyF(ws, stream, down1, x0, B, 32, 32, 64, 64);
    float* x1 = (float*)ws.P((size_t)B * 32 * 32 * 32 * 4);
    convop(ws, stream, s1, down1.pool_w, down1.pool_b, x1, B, 32, 64, 64, 32, 3, 2, 1, FB);

    float* s2 = stageBodyF(ws, stream, down2, x1, B, 32, 64, 32, 32);
    float* x2 = (float*)ws.P((size_t)B * 64 * 16 * 16 * 4);
    convop(ws, stream, s2, down2.pool_w, down2.pool_b, x2, B, 64, 32, 32, 64, 3, 2, 1, FB);

    float* s3 = stageBodyF(ws, stream, down3, x2, B, 64, 128, 16, 16);
    float* x3 = (float*)ws.P((size_t)B * 128 * 8 * 8 * 4);
    convop(ws, stream, s3, down3.pool_w, down3.pool_b, x3, B, 128, 16, 16, 128, 3, 2, 1, FB);

    // ---- mid ----
    float* m = stageBodyF(ws, stream, midS, x3, B, 128, 256, 8, 8);

    // ---- decoder ----
    float* u3 = upStageF(ws, stream, up3S, m,  s3, B, 256, 128, 8, 8);
    float* u2 = upStageF(ws, stream, up2S, u3, s2, B, 128, 64, 16, 16);
    float* u1 = upStageF(ws, stream, up1S, u2, s1, B, 64, 32, 32, 32);

    // ---- head: two conv-transpose(2,2) ----
    float* o = (float*)ws.P((size_t)B * 16 * 128 * 128 * 4);
    k_convt2<<<gdu((long)B * 16 * 128 * 128), TPB, 0, stream>>>(
        u1, out0_w, out0_b, o, 32, 16, 64, 64, B, 1);
    k_convt2<<<gdu((long)B * 1 * 256 * 256), TPB, 0, stream>>>(
        o, out1_w, out1_b, (float*)d_out, 16, 1, 128, 128, B, 0);
}
